// CausalSelfAttention_18305150616211
// MI455X (gfx1250) — matrix operand order repaired and checked by reading, not compiled, not run on hardware
//
#include <hip/hip_runtime.h>
#include <hip/hip_bf16.h>
#include <stdint.h>

// ---- CDNA5 types ------------------------------------------------------------
typedef __attribute__((ext_vector_type(16))) __bf16 v16bf;
typedef __attribute__((ext_vector_type(8)))  __bf16 v8bf;
typedef __attribute__((ext_vector_type(8)))  float  v8f;
typedef __attribute__((ext_vector_type(4)))  unsigned int v4u;
typedef __attribute__((ext_vector_type(8)))  int v8i;
typedef __attribute__((ext_vector_type(4)))  int v4i;

#define AS1 __attribute__((address_space(1)))
#define AS3 __attribute__((address_space(3)))

#if __has_builtin(__builtin_amdgcn_global_load_async_to_lds_b128)
#define HAVE_ASYNC_LDS 1
#else
#define HAVE_ASYNC_LDS 0
#endif
#if __has_builtin(__builtin_amdgcn_tensor_load_to_lds)
#define HAVE_TDM 1
#else
#define HAVE_TDM 0
#endif

#define Bn 4
#define Tn 2048
#define Cn 1024
#define Hn 16
#define Dn 64

__device__ __forceinline__ unsigned short f32_to_bf16_rne(float f) {
  union { float f; unsigned int u; } v; v.f = f;
  unsigned int r = v.u + 0x7FFFu + ((v.u >> 16) & 1u);
  return (unsigned short)(r >> 16);
}

// Async 16B global -> LDS (no VGPR round trip, ASYNCcnt-tracked).
// Builtin prototype (from toolchain diagnostics): (v4i AS1*, v4i AS3*, imm, imm).
__device__ __forceinline__ void async_ld_b128(void* l, const void* g) {
#if HAVE_ASYNC_LDS
  __builtin_amdgcn_global_load_async_to_lds_b128(
      (AS1 v4i*)(unsigned long long)(uintptr_t)g,
      (AS3 v4i*)(unsigned)(uintptr_t)l, 0, 0);
#else
  *(v8bf*)l = *(const v8bf*)g;                 // sync fallback
#endif
}

__device__ __forceinline__ void wait_async0() {
#if HAVE_ASYNC_LDS
#if __has_builtin(__builtin_amdgcn_s_wait_asynccnt)
  __builtin_amdgcn_s_wait_asynccnt(0);
#else
  asm volatile("s_wait_asynccnt 0x0" ::: "memory");
#endif
#endif
}

__device__ __forceinline__ void wait_tensor0() {
#if HAVE_TDM
#if __has_builtin(__builtin_amdgcn_s_wait_tensorcnt)
  __builtin_amdgcn_s_wait_tensorcnt(0);
#else
  asm volatile("s_wait_tensorcnt 0x0" ::: "memory");
#endif
#endif
}

// TDM: move a 2D bf16 tile (tile_d0 x tile_d1 elements, row stride stride0
// elements) from global memory into LDS with one descriptor. Issued per-wave.
__device__ __forceinline__ void tdm_load_2d_bf16(void* lds_dst, const void* gsrc,
                                                 unsigned tile_d0, unsigned tile_d1,
                                                 unsigned tensor_d0, unsigned tensor_d1,
                                                 unsigned long long stride0) {
#if HAVE_TDM
  unsigned long long ga = (unsigned long long)(uintptr_t)gsrc;
  unsigned lds_off = (unsigned)(uintptr_t)lds_dst;  // flat LDS addr low bits == LDS offset
  v4u g0;
  g0[0] = 1u;                                         // count=1, user descriptor
  g0[1] = lds_off;                                    // lds_addr [63:32]
  g0[2] = (unsigned)(ga & 0xFFFFFFFFu);               // global_addr lo
  g0[3] = (unsigned)((ga >> 32) & 0x01FFFFFFu) | (2u << 30);  // global hi | type=2
  v8i g1;
  g1[0] = (int)(1u << 16);                            // data_size=1 (2 bytes)
  g1[1] = (int)((tensor_d0 & 0xFFFFu) << 16);         // tensor_dim0 lo16 (bits 63:48)
  g1[2] = (int)((tensor_d0 >> 16) & 0xFFFFu) |        // tensor_dim0 hi16
          (int)((tensor_d1 & 0xFFFFu) << 16);         // tensor_dim1 lo16
  g1[3] = (int)((tensor_d1 >> 16) & 0xFFFFu) |        // tensor_dim1 hi16
          (int)((tile_d0 & 0xFFFFu) << 16);           // tile_dim0
  g1[4] = (int)(tile_d1 & 0xFFFFu);                   // tile_dim1 (tile_dim2=0)
  g1[5] = (int)(unsigned)(stride0 & 0xFFFFFFFFu);     // tensor_dim0_stride lo32
  g1[6] = (int)(unsigned)((stride0 >> 32) & 0xFFFFu); // stride hi16 (stride1=0)
  g1[7] = 0;
  v4i z = {};
#if defined(__clang_major__) && (__clang_major__ >= 23)
  v8i z8 = {};
  __builtin_amdgcn_tensor_load_to_lds(g0, g1, z, z, z8, 0);
#else
  __builtin_amdgcn_tensor_load_to_lds(g0, g1, z, z, 0);
#endif
#else
  (void)lds_dst; (void)gsrc; (void)tile_d0; (void)tile_d1;
  (void)tensor_d0; (void)tensor_d1; (void)stride0;
#endif
}

// Build a 16x32 bf16 A-fragment for this lane.
// p = row base (row = m0 + (lane&15)) already offset by k0 + (lane>>4)*8.
__device__ __forceinline__ v16bf make_a_frag(const unsigned short* p) {
  union { v16bf v; v8bf h[2]; } u;
  u.h[0] = *(const v8bf*)(p);
  u.h[1] = *(const v8bf*)(p + 16);
  return u.v;
}

// ---- small utility kernels --------------------------------------------------
__global__ void k_cvt_bf16(const float* __restrict__ src,
                           unsigned short* __restrict__ dst, int n) {
  int i = blockIdx.x * blockDim.x + threadIdx.x;
  if (i < n) dst[i] = f32_to_bf16_rne(src[i]);
}

__global__ void k_maskbias(const int* __restrict__ mask,
                           float* __restrict__ mb, int n) {
  int i = blockIdx.x * blockDim.x + threadIdx.x;
  if (i < n) mb[i] = (mask[i] != 0) ? 0.0f : -__builtin_inff();
}

// ---- QKV GEMM: [8192,1024] x [1024,3072] + bias -> Q/K/V [B,H,T,D] bf16 ----
// Double-buffered TDM tile staging pipelined against WMMA compute.
__global__ __launch_bounds__(256) void k_qkv_gemm(
    const unsigned short* __restrict__ Ag,   // x bf16 [8192,1024]
    const unsigned short* __restrict__ Bg,   // Wqkv bf16 [1024,3072]
    const float* __restrict__ bias,          // [3072] f32
    unsigned short* __restrict__ Qb,
    unsigned short* __restrict__ Kb,
    unsigned short* __restrict__ Vb) {
  __shared__ unsigned short As[2][128 * 32];
  __shared__ unsigned short Bs[2][32 * 128];
  const int tid  = threadIdx.x;
  const int lane = tid & 31, wave = tid >> 5;
  const int half = lane >> 4, lc = lane & 15;
  const int m0 = blockIdx.x * 128;
  const int n0 = blockIdx.y * 128;
  const int wm = (wave & 1) * 64;
  const int wn = (wave >> 1) * 32;

  auto stage = [&](int kt, int buf) {
#if HAVE_TDM
    if (wave == 0) {
      tdm_load_2d_bf16(&As[buf][0], Ag + (size_t)m0 * Cn + kt, 32, 128, 32, 128, Cn);
      tdm_load_2d_bf16(&Bs[buf][0], Bg + (size_t)kt * 3072 + n0, 128, 32, 128, 32, 3072ull);
    }
#else
#pragma unroll
    for (int i = tid; i < 512; i += 256) {
      int r = i >> 2, c = (i & 3) << 3;
      async_ld_b128(&As[buf][r * 32 + c], Ag + (size_t)(m0 + r) * Cn + kt + c);
    }
#pragma unroll
    for (int i = tid; i < 512; i += 256) {
      int r = i >> 4, c = (i & 15) << 3;
      async_ld_b128(&Bs[buf][r * 128 + c], Bg + (size_t)(kt + r) * 3072 + n0 + c);
    }
#endif
  };
  auto stage_wait = [&]() {
#if HAVE_TDM
    if (wave == 0) wait_tensor0();
#else
    wait_async0();
#endif
  };

  v8f zero = {};
  v8f acc[4][2];
#pragma unroll
  for (int mi = 0; mi < 4; ++mi)
#pragma unroll
    for (int ni = 0; ni < 2; ++ni) acc[mi][ni] = zero;

  stage(0, 0);
  stage_wait();
  __syncthreads();

  for (int kt = 0; kt < Cn; kt += 32) {
    const int buf = (kt >> 5) & 1;
    if (kt + 32 < Cn) stage(kt + 32, buf ^ 1);     // prefetch next tile via DMA
    if (kt + 64 < Cn) {                            // L2 warm 2 tiles ahead
      __builtin_prefetch(Ag + (size_t)(m0 + (tid >> 1)) * Cn + kt + 64, 0, 1);
      __builtin_prefetch(Bg + (size_t)(kt + 64 + (tid >> 3)) * 3072 + n0, 0, 1);
    }

    v16bf af[4], bf[2];
#pragma unroll
    for (int mi = 0; mi < 4; ++mi)
      af[mi] = make_a_frag(&As[buf][(wm + mi * 16 + lc) * 32 + half * 8]);
#pragma unroll
    for (int ni = 0; ni < 2; ++ni) {
      union { v16bf v; unsigned short u[16]; } tb;
#pragma unroll
      for (int s = 0; s < 16; ++s)
        tb.u[s] = Bs[buf][(half * 8 + (s & 7) + ((s >> 3) << 4)) * 128 + wn + ni * 16 + lc];
      bf[ni] = tb.v;
    }
#pragma unroll
    for (int mi = 0; mi < 4; ++mi)
#pragma unroll
      for (int ni = 0; ni < 2; ++ni)
        acc[mi][ni] = __builtin_amdgcn_wmma_f32_16x16x32_bf16(
            false, af[mi], false, bf[ni], (short)0, acc[mi][ni], false, false);

    stage_wait();                                  // next tile landed
    __syncthreads();                               // all waves done reading buf
  }

  // Epilogue: bias + scatter into [B,H,T,D] bf16 layouts.
#pragma unroll
  for (int mi = 0; mi < 4; ++mi) {
#pragma unroll
    for (int ni = 0; ni < 2; ++ni) {
#pragma unroll
      for (int r = 0; r < 8; ++r) {
        int R  = m0 + wm + mi * 16 + r + half * 8;   // row in [0,8192)
        int Cc = n0 + wn + ni * 16 + lc;             // col in [0,3072)
        float v = acc[mi][ni][r] + bias[Cc];
        int which = Cc >> 10;
        int cc = Cc & 1023;
        int h = cc >> 6, d = cc & 63;
        int b = R >> 11, t = R & 2047;
        size_t dst = (((size_t)(b * Hn + h)) * Tn + t) * Dn + d;
        unsigned short bv = f32_to_bf16_rne(v);
        if (which == 0)      Qb[dst] = bv;
        else if (which == 1) Kb[dst] = bv;
        else                 Vb[dst] = bv;
      }
    }
  }
}

// ---- Flash attention: per (b,h) 64-row q tile, stream 64-row kv tiles ------
// K via async-to-LDS (read as [t][d] rows), V via TDM; both double-buffered.
__global__ __launch_bounds__(128) void k_attn(
    const unsigned short* __restrict__ Qg,   // [B,H,T,D] bf16
    const unsigned short* __restrict__ Kg,
    const unsigned short* __restrict__ Vg,
    const float* __restrict__ maskbias,      // [B,T] 0 or -inf
    unsigned short* __restrict__ Yb) {       // [B,T,C] bf16
  __shared__ unsigned short Ks[2][64 * 64];  // [t][d] row-major K tiles
  __shared__ unsigned short Vs[2][64 * 64];  // [t][d] V tiles
  __shared__ unsigned short Kt[64 * 64];     // [d][t] transposed K tile
  __shared__ unsigned short Ps[4 * 16 * 64]; // per-wave P staging
  const int tid  = threadIdx.x;
  const int lane = tid & 31, wave = tid >> 5;
  const int half = lane >> 4, lc = lane & 15;
  const int qt = blockIdx.x;
  const int q0 = qt * 64;
  const int bh = blockIdx.y;
  const int b = bh >> 4, h = bh & 15;
  const size_t base = (size_t)bh * Tn * Dn;

  auto stage_kv = [&](int jj, int buf) {
    const int tt0 = jj * 64;
#pragma unroll
    for (int i = tid; i < 512; i += 128) {
      int tt = i >> 3, c = (i & 7) << 3;
      async_ld_b128(&Ks[buf][tt * 64 + c], Kg + base + (size_t)(tt0 + tt) * Dn + c);
    }
#if HAVE_TDM
    if (wave == 0)
      tdm_load_2d_bf16(&Vs[buf][0], Vg + base + (size_t)tt0 * Dn, 64, 64, 64, 64, 64ull);
#else
#pragma unroll
    for (int i = tid; i < 512; i += 128) {
      int tt = i >> 3, c = (i & 7) << 3;
      async_ld_b128(&Vs[buf][tt * 64 + c], Vg + base + (size_t)(tt0 + tt) * Dn + c);
    }
#endif
  };
  auto stage_wait = [&]() {
    wait_async0();
#if HAVE_TDM
    if (wave == 0) wait_tensor0();
#endif
  };

  // Q A-fragments live in registers for the whole kv loop.
  const unsigned short* qrow = Qg + base + (size_t)(q0 + wave * 16 + lc) * Dn;
  v16bf aq[2];
#pragma unroll
  for (int kd = 0; kd < 2; ++kd) aq[kd] = make_a_frag(qrow + kd * 32 + half * 8);

  v8f zero = {};
  float mrun[8], lrun[8];
  v8f accO[4];
#pragma unroll
  for (int r = 0; r < 8; ++r) { mrun[r] = -3.0e38f; lrun[r] = 0.0f; }
#pragma unroll
  for (int d = 0; d < 4; ++d) accO[d] = zero;

  const float* mb = maskbias + b * Tn;

  stage_kv(0, 0);
  stage_wait();
  __syncthreads();

  for (int j = 0; j <= qt; ++j) {
    const int t0 = j * 64;
    const int buf = j & 1;
    if (j < qt) stage_kv(j + 1, buf ^ 1);          // pipeline next kv tile

    // In-LDS transpose Ks[buf][t][d] -> Kt[d][t] (Kt is unused; B operand reads Ks rows, lane<->t).
#pragma unroll
    for (int i = tid; i < 512; i += 128) {
      int tt = i >> 3, c = (i & 7) << 3;
      v8bf kv = *(const v8bf*)(&Ks[buf][tt * 64 + c]);
#pragma unroll
      for (int e = 0; e < 8; ++e)
        Kt[(c + e) * 64 + tt] = ((const unsigned short*)&kv)[e];
    }
    __syncthreads();

    // S = Q * K^T  (contraction over D=64 -> two k-steps of 32)
    v8f s[4];
#pragma unroll
    for (int n = 0; n < 4; ++n) {
      s[n] = zero;
#pragma unroll
      for (int kd = 0; kd < 2; ++kd) {
        v16bf bk = make_a_frag(&Ks[buf][(n * 16 + lc) * 64 + kd * 32 + half * 8]);
        s[n] = __builtin_amdgcn_wmma_f32_16x16x32_bf16(
            false, aq[kd], false, bk, (short)0, s[n], false, false);
      }
    }

    // scale, padding bias, causal mask, row max of this tile
    float rmax[8];
#pragma unroll
    for (int r = 0; r < 8; ++r) rmax[r] = -3.0e38f;
#pragma unroll
    for (int n = 0; n < 4; ++n) {
      int   kj = t0 + n * 16 + lc;
      float pb = mb[kj];
#pragma unroll
      for (int r = 0; r < 8; ++r) {
        float x = s[n][r] * 0.125f + pb;             // 1/sqrt(64)
        if (j == qt) {
          int qi = q0 + wave * 16 + r + half * 8;
          if (kj > qi) x = -__builtin_inff();
        }
        s[n][r] = x;
        rmax[r] = fmaxf(rmax[r], x);
      }
    }
#pragma unroll
    for (int r = 0; r < 8; ++r) {                    // row reduce within half
      float v = rmax[r];
      v = fmaxf(v, __shfl_xor(v, 1, 32));
      v = fmaxf(v, __shfl_xor(v, 2, 32));
      v = fmaxf(v, __shfl_xor(v, 4, 32));
      v = fmaxf(v, __shfl_xor(v, 8, 32));
      float mnew  = fmaxf(mrun[r], v);
      float alpha = __expf(mrun[r] - mnew);
      mrun[r] = mnew;
      lrun[r] *= alpha;
#pragma unroll
      for (int d = 0; d < 4; ++d) accO[d][r] *= alpha;
      rmax[r] = mnew;
    }

    // P = exp(S - m): row sums + stage to LDS (bf16) for A-fragment reuse
    float rsum[8];
#pragma unroll
    for (int r = 0; r < 8; ++r) rsum[r] = 0.0f;
#pragma unroll
    for (int n = 0; n < 4; ++n) {
#pragma unroll
      for (int r = 0; r < 8; ++r) {
        float p = __expf(s[n][r] - rmax[r]);
        rsum[r] += p;
        Ps[wave * 1024 + (r + half * 8) * 64 + n * 16 + lc] = f32_to_bf16_rne(p);
      }
    }
#pragma unroll
    for (int r = 0; r < 8; ++r) {
      float v = rsum[r];
      v += __shfl_xor(v, 1, 32);
      v += __shfl_xor(v, 2, 32);
      v += __shfl_xor(v, 4, 32);
      v += __shfl_xor(v, 8, 32);
      lrun[r] += v;
    }
    __syncthreads();

    // O += P * V
    const unsigned short* prow = Ps + wave * 1024 + lc * 64;
    v16bf ap[2];
#pragma unroll
    for (int kd = 0; kd < 2; ++kd) ap[kd] = make_a_frag(prow + kd * 32 + half * 8);
#pragma unroll
    for (int d = 0; d < 4; ++d)
#pragma unroll
      for (int kd = 0; kd < 2; ++kd) {
        union { v16bf v; unsigned short u[16]; } tv;
#pragma unroll
        for (int s = 0; s < 16; ++s)
          tv.u[s] = Vs[buf][(kd * 32 + half * 8 + (s & 7) + ((s >> 3) << 4)) * 64 + d * 16 + lc];
        v16bf bv = tv.v;
        accO[d] = __builtin_amdgcn_wmma_f32_16x16x32_bf16(
            false, ap[kd], false, bv, (short)0, accO[d], false, false);
      }

    stage_wait();                                  // next kv tile landed
    __syncthreads();                               // all waves done with buf
  }

  // normalize and write y in [B,T,H*D] bf16 (row-major for proj GEMM)
#pragma unroll
  for (int r = 0; r < 8; ++r) {
    float inv = 1.0f / lrun[r];
    int t = q0 + wave * 16 + r + half * 8;
    size_t row = ((size_t)b * Tn + t) * Cn + h * Dn;
#pragma unroll
    for (int d = 0; d < 4; ++d)
      Yb[row + d * 16 + lc] = f32_to_bf16_rne(accO[d][r] * inv);
  }
}

// ---- Proj GEMM: [8192,1024] x [1024,1024] + bias -> f32 out ---------------
__global__ __launch_bounds__(256) void k_proj_gemm(
    const unsigned short* __restrict__ Ag,   // y bf16 [8192,1024]
    const unsigned short* __restrict__ Bg,   // Wproj bf16 [1024,1024]
    const float* __restrict__ bias,          // [1024]
    float* __restrict__ out) {               // [8192,1024] f32
  __shared__ unsigned short As[2][128 * 32];
  __shared__ unsigned short Bs[2][32 * 128];
  const int tid  = threadIdx.x;
  const int lane = tid & 31, wave = tid >> 5;
  const int half = lane >> 4, lc = lane & 15;
  const int m0 = blockIdx.x * 128;
  const int n0 = blockIdx.y * 128;
  const int wm = (wave & 1) * 64;
  const int wn = (wave >> 1) * 32;

  auto stage = [&](int kt, int buf) {
#if HAVE_TDM
    if (wave == 0) {
      tdm_load_2d_bf16(&As[buf][0], Ag + (size_t)m0 * Cn + kt, 32, 128, 32, 128, Cn);
      tdm_load_2d_bf16(&Bs[buf][0], Bg + (size_t)kt * Cn + n0, 128, 32, 128, 32, Cn);
    }
#else
#pragma unroll
    for (int i = tid; i < 512; i += 256) {
      int r = i >> 2, c = (i & 3) << 3;
      async_ld_b128(&As[buf][r * 32 + c], Ag + (size_t)(m0 + r) * Cn + kt + c);
    }
#pragma unroll
    for (int i = tid; i < 512; i += 256) {
      int r = i >> 4, c = (i & 15) << 3;
      async_ld_b128(&Bs[buf][r * 128 + c], Bg + (size_t)(kt + r) * Cn + n0 + c);
    }
#endif
  };
  auto stage_wait = [&]() {
#if HAVE_TDM
    if (wave == 0) wait_tensor0();
#else
    wait_async0();
#endif
  };

  v8f zero = {};
  v8f acc[4][2];
#pragma unroll
  for (int mi = 0; mi < 4; ++mi)
#pragma unroll
    for (int ni = 0; ni < 2; ++ni) acc[mi][ni] = zero;

  stage(0, 0);
  stage_wait();
  __syncthreads();

  for (int kt = 0; kt < Cn; kt += 32) {
    const int buf = (kt >> 5) & 1;
    if (kt + 32 < Cn) stage(kt + 32, buf ^ 1);
    if (kt + 64 < Cn) {
      __builtin_prefetch(Ag + (size_t)(m0 + (tid >> 1)) * Cn + kt + 64, 0, 1);
      __builtin_prefetch(Bg + (size_t)(kt + 64 + (tid >> 3)) * Cn + n0, 0, 1);
    }

    v16bf af[4], bf[2];
#pragma unroll
    for (int mi = 0; mi < 4; ++mi)
      af[mi] = make_a_frag(&As[buf][(wm + mi * 16 + lc) * 32 + half * 8]);
#pragma unroll
    for (int ni = 0; ni < 2; ++ni) {
      union { v16bf v; unsigned short u[16]; } tb;
#pragma unroll
      for (int s = 0; s < 16; ++s)
        tb.u[s] = Bs[buf][(half * 8 + (s & 7) + ((s >> 3) << 4)) * 128 + wn + ni * 16 + lc];
      bf[ni] = tb.v;
    }
#pragma unroll
    for (int mi = 0; mi < 4; ++mi)
#pragma unroll
      for (int ni = 0; ni < 2; ++ni)
        acc[mi][ni] = __builtin_amdgcn_wmma_f32_16x16x32_bf16(
            false, af[mi], false, bf[ni], (short)0, acc[mi][ni], false, false);

    stage_wait();
    __syncthreads();
  }

#pragma unroll
  for (int mi = 0; mi < 4; ++mi)
#pragma unroll
    for (int ni = 0; ni < 2; ++ni)
#pragma unroll
      for (int r = 0; r < 8; ++r) {
        int R  = m0 + wm + mi * 16 + r + half * 8;
        int Cc = n0 + wn + ni * 16 + lc;
        out[(size_t)R * Cn + Cc] = acc[mi][ni][r] + bias[Cc];
      }
}

// ---- host-side launcher -----------------------------------------------------
extern "C" void kernel_launch(void* const* d_in, const int* in_sizes, int n_in,
                              void* d_out, int out_size, void* d_ws, size_t ws_size,
                              hipStream_t stream) {
  (void)in_sizes; (void)n_in; (void)out_size; (void)ws_size;
  const float* x     = (const float*)d_in[0];
  const int*   pmask = (const int*)d_in[1];
  const float* Wqkv  = (const float*)d_in[2];
  const float* bqkv  = (const float*)d_in[3];
  const float* Wproj = (const float*)d_in[4];
  const float* bproj = (const float*)d_in[5];
  float* out = (float*)d_out;

  char* w = (char*)d_ws;
  size_t off = 0;
  auto alloc = [&](size_t bytes) -> void* {
    void* p = w + off;
    off = (off + bytes + 255) & ~(size_t)255;
    return p;
  };
  const size_t nX  = (size_t)Bn * Tn * Cn;        // 8,388,608
  unsigned short* xb  = (unsigned short*)alloc(nX * 2);
  unsigned short* wqb = (unsigned short*)alloc((size_t)Cn * 3 * Cn * 2);
  unsigned short* wpb = (unsigned short*)alloc((size_t)Cn * Cn * 2);
  unsigned short* Qb  = (unsigned short*)alloc(nX * 2);
  unsigned short* Kb  = (unsigned short*)alloc(nX * 2);
  unsigned short* Vb  = (unsigned short*)alloc(nX * 2);
  unsigned short* Yb  = (unsigned short*)alloc(nX * 2);
  float*          mb  = (float*)alloc((size_t)Bn * Tn * 4);

  const int nXi = (int)nX, nWq = Cn * 3 * Cn, nWp = Cn * Cn, nM = Bn * Tn;
  k_cvt_bf16<<<(nXi + 255) / 256, 256, 0, stream>>>(x, xb, nXi);
  k_cvt_bf16<<<(nWq + 255) / 256, 256, 0, stream>>>(Wqkv, wqb, nWq);
  k_cvt_bf16<<<(nWp + 255) / 256, 256, 0, stream>>>(Wproj, wpb, nWp);
  k_maskbias<<<(nM + 255) / 256, 256, 0, stream>>>(pmask, mb, nM);

  k_qkv_gemm<<<dim3(64, 24), 256, 0, stream>>>(xb, wqb, bqkv, Qb, Kb, Vb);
  k_attn<<<dim3(Tn / 64, Bn * Hn), 128, 0, stream>>>(Qb, Kb, Vb, mb, Yb);
  k_proj_gemm<<<dim3(64, 8), 256, 0, stream>>>(Yb, wpb, bproj, out);
}
